// GCN_34445637714219
// MI455X (gfx1250) — compile-verified, matrix operand order checked by reading, not run on hardware
//
#include <hip/hip_runtime.h>
#include <math.h>

// ---------------- problem constants (match reference) ----------------
#define NNODES 100000
#define NEDGES 1600000
#define FIN    256
#define HID    64
#define NCLS   40
#define CPAD   48                  // NCLS padded to 3 WMMA N-tiles
#define MTILES (NNODES / 16)       // 6250 exactly

typedef __attribute__((ext_vector_type(16))) __bf16 v16bf;
typedef __attribute__((ext_vector_type(8)))  float  v8f;

struct BF16Frag { uint4 lo, hi; };   // 32 bytes = 16 bf16 = one WMMA A/B fragment

static __device__ __forceinline__ v16bf frag_cast(BF16Frag f) {
    return __builtin_bit_cast(v16bf, f);
}

// round-to-nearest-even f32 -> bf16
static __device__ __forceinline__ unsigned short f2bf(float f) {
    unsigned u = __builtin_bit_cast(unsigned, f);
    u += 0x7FFFu + ((u >> 16) & 1u);
    return (unsigned short)(u >> 16);
}

// ---------------- degree / normalization ----------------
__global__ void deg_init_kernel(float* __restrict__ dis) {
    int i = blockIdx.x * blockDim.x + threadIdx.x;
    if (i < NNODES) dis[i] = 1.0f;              // self-loop contributes 1
}

__global__ void deg_accum_kernel(const int* __restrict__ col, float* __restrict__ dis) {
    int e = blockIdx.x * blockDim.x + threadIdx.x;
    if (e < NEDGES) atomicAdd(&dis[col[e]], 1.0f);
}

__global__ void deg_rsqrt_kernel(float* __restrict__ dis) {
    int i = blockIdx.x * blockDim.x + threadIdx.x;
    if (i < NNODES) dis[i] = rsqrtf(dis[i]);    // deg >= 1 always (self-loops)
}

// ---------------- fp32 -> bf16 conversions ----------------
__global__ void cvt_x_kernel(const float* __restrict__ x, unsigned short* __restrict__ xbf) {
    int t = blockIdx.x * blockDim.x + threadIdx.x;   // N*FIN/4 threads
    if (t >= NNODES * (FIN / 4)) return;
    const float4 v = ((const float4*)x)[t];
    uint2 o;
    o.x = (unsigned)f2bf(v.x) | ((unsigned)f2bf(v.y) << 16);
    o.y = (unsigned)f2bf(v.z) | ((unsigned)f2bf(v.w) << 16);
    ((uint2*)xbf)[t] = o;
}

__global__ void cvt_w1t_kernel(const float* __restrict__ W1, unsigned short* __restrict__ w1t) {
    int t = blockIdx.x * blockDim.x + threadIdx.x;   // HID*FIN threads
    if (t >= HID * FIN) return;
    int n = t >> 8;          // FIN == 256
    int k = t & 255;
    w1t[t] = f2bf(W1[k * HID + n]);
}

__global__ void cvt_w2t_kernel(const float* __restrict__ W2, unsigned short* __restrict__ w2t) {
    int t = blockIdx.x * blockDim.x + threadIdx.x;   // CPAD*HID threads
    if (t >= CPAD * HID) return;
    int n = t >> 6;          // HID == 64
    int k = t & 63;
    w2t[t] = (n < NCLS) ? f2bf(W2[k * NCLS + n]) : (unsigned short)0;
}

// ---------------- GEMM1: h[N,64] = x[N,256] @ W1  (bf16 WMMA, f32 acc) ----------------
__global__ __launch_bounds__(128)
void gemm1_kernel(const unsigned short* __restrict__ xbf,
                  const unsigned short* __restrict__ w1t,
                  float* __restrict__ h) {
    int mtile = blockIdx.x * 4 + (threadIdx.x >> 5);
    if (mtile >= MTILES) return;
    int lane = threadIdx.x & 31;
    int r    = lane & 15;        // row within tile (A) / col within tile (B)
    int half = lane >> 4;        // which K-half this lane holds
    const unsigned short* arow = xbf + (size_t)(mtile * 16 + r) * FIN;

    v8f acc[4] = {};
    #pragma unroll
    for (int kk = 0; kk < FIN / 32; ++kk) {
        const unsigned short* ap = arow + kk * 32 + half * 8;
        __builtin_prefetch(ap + 32, 0, 0);                 // global_prefetch_b8
        BF16Frag af;
        af.lo = *(const uint4*)(ap);
        af.hi = *(const uint4*)(ap + 16);
        v16bf av = frag_cast(af);
        #pragma unroll
        for (int nt = 0; nt < 4; ++nt) {
            const unsigned short* bp = w1t + (size_t)(nt * 16 + r) * FIN + kk * 32 + half * 8;
            BF16Frag bf_;
            bf_.lo = *(const uint4*)(bp);
            bf_.hi = *(const uint4*)(bp + 16);
            acc[nt] = __builtin_amdgcn_wmma_f32_16x16x32_bf16(
                false, av, false, frag_cast(bf_), (short)0, acc[nt], false, false);
        }
    }
    #pragma unroll
    for (int nt = 0; nt < 4; ++nt)
        #pragma unroll
        for (int j = 0; j < 8; ++j) {
            int m = mtile * 16 + j + 8 * half;
            h[(size_t)m * HID + nt * 16 + r] = acc[nt][j];
        }
}

// ---------------- layer-1 aggregation ----------------
// acc1[i,f] = b1[f] + dis[i]^2 * h[i,f]   (self-loop term + bias)
__global__ void acc1_init_kernel(const float* __restrict__ h, const float* __restrict__ dis,
                                 const float* __restrict__ b1, float* __restrict__ acc1) {
    int t = blockIdx.x * blockDim.x + threadIdx.x;   // N*HID threads
    if (t >= NNODES * HID) return;
    int i = t >> 6, f = t & 63;
    float d = dis[i];
    acc1[t] = b1[f] + d * d * h[t];
}

// acc1[col,f] += dis[row]*dis[col] * h[row,f]
__global__ void scatter1_kernel(const int* __restrict__ row, const int* __restrict__ col,
                                const float* __restrict__ dis, const float* __restrict__ h,
                                float* __restrict__ acc1) {
    int t = blockIdx.x * blockDim.x + threadIdx.x;   // NEDGES*64 threads
    int e = t >> 6, f = t & 63;
    if (e >= NEDGES) return;
    int rr = row[e], cc = col[e];
    float nrm = dis[rr] * dis[cc];
    atomicAdd(&acc1[(size_t)cc * HID + f], nrm * h[(size_t)rr * HID + f]);
}

// h2bf = bf16(relu(acc1))
__global__ void relu_cvt_kernel(const float* __restrict__ acc1, unsigned short* __restrict__ h2bf) {
    int t = blockIdx.x * blockDim.x + threadIdx.x;
    if (t >= NNODES * HID) return;
    h2bf[t] = f2bf(fmaxf(acc1[t], 0.0f));
}

// ---------------- GEMM2: z[N,48] = h2[N,64] @ W2pad  (bf16 WMMA) ----------------
__global__ __launch_bounds__(128)
void gemm2_kernel(const unsigned short* __restrict__ h2bf,
                  const unsigned short* __restrict__ w2t,
                  float* __restrict__ z) {
    int mtile = blockIdx.x * 4 + (threadIdx.x >> 5);
    if (mtile >= MTILES) return;
    int lane = threadIdx.x & 31;
    int r    = lane & 15;
    int half = lane >> 4;
    const unsigned short* arow = h2bf + (size_t)(mtile * 16 + r) * HID;

    v8f acc[3] = {};
    #pragma unroll
    for (int kk = 0; kk < HID / 32; ++kk) {
        const unsigned short* ap = arow + kk * 32 + half * 8;
        BF16Frag af;
        af.lo = *(const uint4*)(ap);
        af.hi = *(const uint4*)(ap + 16);
        v16bf av = frag_cast(af);
        #pragma unroll
        for (int nt = 0; nt < 3; ++nt) {
            const unsigned short* bp = w2t + (size_t)(nt * 16 + r) * HID + kk * 32 + half * 8;
            BF16Frag bf_;
            bf_.lo = *(const uint4*)(bp);
            bf_.hi = *(const uint4*)(bp + 16);
            acc[nt] = __builtin_amdgcn_wmma_f32_16x16x32_bf16(
                false, av, false, frag_cast(bf_), (short)0, acc[nt], false, false);
        }
    }
    #pragma unroll
    for (int nt = 0; nt < 3; ++nt)
        #pragma unroll
        for (int j = 0; j < 8; ++j) {
            int m = mtile * 16 + j + 8 * half;
            z[(size_t)m * CPAD + nt * 16 + r] = acc[nt][j];
        }
}

// ---------------- layer-2 aggregation ----------------
__global__ void acc2_init_kernel(const float* __restrict__ z, const float* __restrict__ dis,
                                 const float* __restrict__ b2, float* __restrict__ acc2) {
    int t = blockIdx.x * blockDim.x + threadIdx.x;   // N*NCLS threads
    if (t >= NNODES * NCLS) return;
    int i = t / NCLS, c = t - i * NCLS;
    float d = dis[i];
    acc2[t] = b2[c] + d * d * z[(size_t)i * CPAD + c];
}

__global__ void scatter2_kernel(const int* __restrict__ row, const int* __restrict__ col,
                                const float* __restrict__ dis, const float* __restrict__ z,
                                float* __restrict__ acc2) {
    int t = blockIdx.x * blockDim.x + threadIdx.x;   // NEDGES*64 threads, f<40 active
    int e = t >> 6, f = t & 63;
    if (e >= NEDGES || f >= NCLS) return;
    int rr = row[e], cc = col[e];
    float nrm = dis[rr] * dis[cc];
    atomicAdd(&acc2[(size_t)cc * NCLS + f], nrm * z[(size_t)rr * CPAD + f]);
}

// ---------------- log-softmax over 40 classes ----------------
__global__ void logsoftmax_kernel(const float* __restrict__ acc2, float* __restrict__ out) {
    int i = blockIdx.x * blockDim.x + threadIdx.x;
    if (i >= NNODES) return;
    float v[NCLS];
    float mx = -3.4e38f;
    #pragma unroll
    for (int c = 0; c < NCLS; ++c) { v[c] = acc2[(size_t)i * NCLS + c]; mx = fmaxf(mx, v[c]); }
    float s = 0.0f;
    #pragma unroll
    for (int c = 0; c < NCLS; ++c) s += expf(v[c] - mx);
    float lse = mx + logf(s);
    #pragma unroll
    for (int c = 0; c < NCLS; ++c) out[(size_t)i * NCLS + c] = v[c] - lse;
}

// ---------------- host launcher ----------------
extern "C" void kernel_launch(void* const* d_in, const int* in_sizes, int n_in,
                              void* d_out, int out_size, void* d_ws, size_t ws_size,
                              hipStream_t stream) {
    const float* x     = (const float*)d_in[0];
    const int*   eidx  = (const int*)d_in[1];       // [2, E]: row then col
    const float* W1    = (const float*)d_in[2];
    const float* b1    = (const float*)d_in[3];
    const float* W2    = (const float*)d_in[4];
    const float* b2    = (const float*)d_in[5];
    const int* row = eidx;
    const int* col = eidx + NEDGES;
    float* out = (float*)d_out;

    // workspace carve-out (256B aligned slices)
    char*  base = (char*)d_ws;
    size_t off  = 0;
    auto carve = [&](size_t bytes) -> char* {
        char* p = base + off;
        off = (off + bytes + 255) & ~(size_t)255;
        return p;
    };
    float*          dis  = (float*)         carve((size_t)NNODES * 4);
    unsigned short* xbf  = (unsigned short*)carve((size_t)NNODES * FIN * 2);
    unsigned short* w1t  = (unsigned short*)carve((size_t)HID * FIN * 2);
    unsigned short* w2t  = (unsigned short*)carve((size_t)CPAD * HID * 2);
    float*          h    = (float*)         carve((size_t)NNODES * HID * 4);
    float*          acc1 = (float*)         carve((size_t)NNODES * HID * 4);
    unsigned short* h2bf = (unsigned short*)carve((size_t)NNODES * HID * 2);
    float*          z    = (float*)         carve((size_t)NNODES * CPAD * 4);
    float*          acc2 = (float*)         carve((size_t)NNODES * NCLS * 4);
    (void)ws_size; (void)in_sizes; (void)n_in; (void)out_size;

    const int T = 256;
    // normalization coefficients
    deg_init_kernel  <<<(NNODES + T - 1) / T, T, 0, stream>>>(dis);
    deg_accum_kernel <<<(NEDGES + T - 1) / T, T, 0, stream>>>(col, dis);
    deg_rsqrt_kernel <<<(NNODES + T - 1) / T, T, 0, stream>>>(dis);
    // bf16 conversions
    cvt_x_kernel   <<<(NNODES * (FIN / 4) + T - 1) / T, T, 0, stream>>>(x, xbf);
    cvt_w1t_kernel <<<(HID * FIN + T - 1) / T, T, 0, stream>>>(W1, w1t);
    cvt_w2t_kernel <<<(CPAD * HID + T - 1) / T, T, 0, stream>>>(W2, w2t);
    // layer 1
    gemm1_kernel <<<(MTILES + 3) / 4, 128, 0, stream>>>(xbf, w1t, h);
    acc1_init_kernel <<<(NNODES * HID + T - 1) / T, T, 0, stream>>>(h, dis, b1, acc1);
    scatter1_kernel  <<<(NEDGES * 64) / T, T, 0, stream>>>(row, col, dis, h, acc1);
    relu_cvt_kernel  <<<(NNODES * HID + T - 1) / T, T, 0, stream>>>(acc1, h2bf);
    // layer 2
    gemm2_kernel <<<(MTILES + 3) / 4, 128, 0, stream>>>(h2bf, w2t, z);
    acc2_init_kernel <<<(NNODES * NCLS + T - 1) / T, T, 0, stream>>>(z, dis, b2, acc2);
    scatter2_kernel  <<<(NEDGES * 64) / T, T, 0, stream>>>(row, col, dis, z, acc2);
    // output
    logsoftmax_kernel <<<(NNODES + T - 1) / T, T, 0, stream>>>(acc2, out);
}
